// UberCRSN_27650999452288
// MI455X (gfx1250) — matrix operand order repaired and checked by reading, not compiled, not run on hardware
//
#include <hip/hip_runtime.h>
#include <hip/hip_bf16.h>
#include <math.h>

// ---------------------------------------------------------------------------
// CDNA5 (gfx1250) implementation of the UberCRSN reference.
// bf16 WMMA (v_wmma_f32_16x16x32_bf16) for all GEMMs, fp32 everywhere else.
// ---------------------------------------------------------------------------

typedef __bf16 bf16_t;
typedef __attribute__((ext_vector_type(16))) __bf16 v16bf;
typedef __attribute__((ext_vector_type(8)))  __bf16 v8bf;
typedef __attribute__((ext_vector_type(4)))  __bf16 v4bf;
typedef __attribute__((ext_vector_type(8)))  float  v8f;

#define B_     8
#define S_     128
#define D_     512
#define M_     4
#define DEPTH_ 16
#define THR_   0.9999f
#define EPS_   1e-6f

#define ROWS_   1024            // B*S
#define K2D_    1024            // 2*D
#define COLS_   3072            // 6*D (qr,qi,kr,ki,vr,vi)
#define QKST_   2048            // row stride of qf|kf buffer
#define BSD_    524288          // B*S*D
#define SD_     65536           // S*D

__device__ __forceinline__ v16bf mk16(v8bf lo, v8bf hi) {
    v16bf r;
#pragma unroll
    for (int i = 0; i < 8; ++i) { r[i] = lo[i]; r[8 + i] = hi[i]; }
    return r;
}

__device__ __forceinline__ v8f wmma_bf16(v16bf a, v16bf b, v8f c) {
    // D = A(16x32 bf16) * B(32x16 bf16) + C(16x16 f32)
    return __builtin_amdgcn_wmma_f32_16x16x32_bf16(
        /*neg_a=*/false, a, /*neg_b=*/false, b,
        /*c_mod=*/(short)0, c, /*reuse_a=*/false, /*reuse_b=*/false);
}

__device__ __forceinline__ float sigmoidf_(float x) { return 1.f / (1.f + __expf(-x)); }

// ---------------------------------------------------------------------------
// Pack complex weights into bf16 B-fragment layout (runs once per launch).
// Logical Wbig[m][k in 0..1023][c in 0..3071]; tiles of 32(K) x 16(N),
// per-lane 16 contiguous bf16 (k = 16*(lane/16) + j, contiguous order, matched by the A loads).
// ---------------------------------------------------------------------------
__global__ void k_pack(const float* __restrict__ Wq_r, const float* __restrict__ Wq_i,
                       const float* __restrict__ Wk_r, const float* __restrict__ Wk_i,
                       const float* __restrict__ Wv_r, const float* __restrict__ Wv_i,
                       bf16_t* __restrict__ Wpack) {
    int e = blockIdx.x * 256 + threadIdx.x;              // < 4*32*192*512
    if (e >= M_ * 32 * 192 * 512) return;
    int within = e & 511;
    int tile   = e >> 9;
    int lane = within >> 4, j = within & 15;
    int ct = tile % 192;
    int kt = (tile / 192) & 31;
    int m  = tile / (192 * 32);
    int n = ct * 16 + (lane & 15);                       // output column 0..3071
    int k = kt * 32 + ((lane >> 4) << 4) + j;            // input dim 0..1023
    int region = n >> 9; int ecol = n & 511;
    int kk = k & 511; bool top = k >= 512;               // top half multiplies z_imag
    size_t wi = ((size_t)m * 512 + kk) * 512 + ecol;     // W[m][d=kk][e=ecol]
    float val;
    switch (region) {
        case 0:  val = top ? -Wq_i[wi] : Wq_r[wi]; break;    // qr = zr*Wr - zi*Wi
        case 1:  val = top ?  Wq_r[wi] : Wq_i[wi]; break;    // qi = zr*Wi + zi*Wr
        case 2:  val = top ? -Wk_i[wi] : Wk_r[wi]; break;
        case 3:  val = top ?  Wk_r[wi] : Wk_i[wi]; break;
        case 4:  val = top ? -Wv_i[wi] : Wv_r[wi]; break;
        default: val = top ?  Wv_r[wi] : Wv_i[wi]; break;
    }
    Wpack[e] = (bf16_t)val;
}

// ---------------------------------------------------------------------------
// Init: z = x, build bf16 X=[zr|zi], zero output accumulators / h / cum.
// ---------------------------------------------------------------------------
__global__ void k_init(const float* __restrict__ xr, const float* __restrict__ xi,
                       float* __restrict__ zr, float* __restrict__ zi,
                       bf16_t* __restrict__ Xbf, float* __restrict__ out,
                       float* __restrict__ h, float* __restrict__ cum) {
    int idx = blockIdx.x * 256 + threadIdx.x;            // < 524288
    if (idx >= BSD_) return;
    float a = xr[idx], b = xi[idx];
    zr[idx] = a; zi[idx] = b;
    int row = idx >> 9, d = idx & 511;
    Xbf[(size_t)row * K2D_ + d]        = (bf16_t)a;
    Xbf[(size_t)row * K2D_ + 512 + d]  = (bf16_t)b;
    out[idx] = 0.f; out[BSD_ + idx] = 0.f;
    if (idx < B_ * D_) h[idx] = 0.f;
    if (idx < B_) cum[idx] = 0.f;
}

// pooled[b][c] = mean over S of zr (c<512) / zi (c>=512).  Also = gwp of prev combine.
__global__ void k_pool(const float* __restrict__ zr, const float* __restrict__ zi,
                       float* __restrict__ pooled) {
    int t = blockIdx.x * 256 + threadIdx.x;              // < 8192
    if (t >= B_ * K2D_) return;
    int b = t >> 10, c = t & 1023;
    const float* src = (c < 512) ? zr : zi;
    int d = c & 511;
    float s = 0.f;
    for (int ss = 0; ss < S_; ++ss) s += src[((size_t)b * S_ + ss) * D_ + d];
    pooled[t] = s * (1.f / (float)S_);
}

// GRU input/hidden matmuls: gx = pooled@Wih + bih, gh = h@Whh + bhh
__global__ void k_gru_mm(const float* __restrict__ pooled, const float* __restrict__ h,
                         const float* __restrict__ Wih, const float* __restrict__ Whh,
                         const float* __restrict__ bih, const float* __restrict__ bhh,
                         float* __restrict__ gx, float* __restrict__ gh) {
    int t = blockIdx.x * 256 + threadIdx.x;              // < 24576
    if (t >= 24576) return;
    int which = t / 12288; int r = t % 12288;
    int b = r / 1536, j = r % 1536;
    if (which == 0) {
        float acc = bih[j];
        for (int k = 0; k < K2D_; ++k) acc += pooled[b * K2D_ + k] * Wih[(size_t)k * 1536 + j];
        gx[b * 1536 + j] = acc;
    } else {
        float acc = bhh[j];
        for (int k = 0; k < D_; ++k) acc += h[b * D_ + k] * Whh[(size_t)k * 1536 + j];
        gh[b * 1536 + j] = acc;
    }
}

__global__ void k_gru_gate(const float* __restrict__ gx, const float* __restrict__ gh,
                           float* __restrict__ h) {
    int idx = blockIdx.x * 256 + threadIdx.x;            // < 4096
    if (idx >= B_ * D_) return;
    int b = idx >> 9, d = idx & 511;
    const float* gxb = gx + b * 1536;
    const float* ghb = gh + b * 1536;
    float r = sigmoidf_(gxb[d] + ghb[d]);
    float z = sigmoidf_(gxb[512 + d] + ghb[512 + d]);
    float n = tanhf(gxb[1024 + d] + r * ghb[1024 + d]);
    h[idx] = (1.f - z) * n + z * h[idx];
}

// ---------------------------------------------------------------------------
// QKV GEMM: Y[m][1024][3072] = X[1024][1024] @ Wbig[m].  bf16 WMMA, fp32 acc.
// cols <2048 -> qf|kf (row-major, stride 2048); cols >=2048 -> vT (transposed).
// Workgroup: 256 thr (8 waves) -> 128x64 tile; wave -> 32x32 (2x2 WMMA tiles).
// ---------------------------------------------------------------------------
__global__ void __launch_bounds__(256) k_qkv(const bf16_t* __restrict__ Xbf,
                                             const bf16_t* __restrict__ Wpack,
                                             bf16_t* __restrict__ qk,
                                             bf16_t* __restrict__ vTr,
                                             bf16_t* __restrict__ vTi) {
    __shared__ bf16_t Xs[128 * 32];                      // 8KB staging of X block
    int tid = threadIdx.x;
    int wave = tid >> 5, lane = tid & 31;
    int r15 = lane & 15, hs = lane >> 4;
    int m = blockIdx.z;
    int rowbase = blockIdx.y * 128;
    int colbase = blockIdx.x * 64;
    int wrow = wave & 3, wcol = wave >> 2;

    v8f acc[2][2];
#pragma unroll
    for (int i = 0; i < 2; ++i)
#pragma unroll
        for (int j = 0; j < 2; ++j) acc[i][j] = (v8f){};

    for (int kt = 0; kt < 32; ++kt) {
        __syncthreads();
#pragma unroll
        for (int i = 0; i < 4; ++i) {                    // stage 128x32 bf16
            int L = tid * 4 + i * 1024;
            int row = L >> 5, kk = L & 31;
            *(v4bf*)&Xs[L] = *(const v4bf*)&Xbf[(size_t)(rowbase + row) * K2D_ + kt * 32 + kk];
        }
        __syncthreads();

        v16bf a[2], bfr[2];
#pragma unroll
        for (int i = 0; i < 2; ++i) {
            int arow = wrow * 32 + i * 16 + r15;
            v8bf lo = *(const v8bf*)&Xs[arow * 32 + (hs ? 16 : 0)];
            v8bf hi = *(const v8bf*)&Xs[arow * 32 + (hs ? 24 : 8)];
            a[i] = mk16(lo, hi);
        }
#pragma unroll
        for (int j = 0; j < 2; ++j) {
            int ct = (colbase >> 4) + wcol * 2 + j;
            const bf16_t* tb = Wpack + (((size_t)(m * 32 + kt) * 192 + ct) << 9) + lane * 16;
            if (kt + 1 < 32) __builtin_prefetch(tb + 192 * 512, 0, 1);  // global_prefetch
            bfr[j] = mk16(*(const v8bf*)tb, *(const v8bf*)(tb + 8));
        }
        acc[0][0] = wmma_bf16(a[0], bfr[0], acc[0][0]);
        acc[0][1] = wmma_bf16(a[0], bfr[1], acc[0][1]);
        acc[1][0] = wmma_bf16(a[1], bfr[0], acc[1][0]);
        acc[1][1] = wmma_bf16(a[1], bfr[1], acc[1][1]);
    }

#pragma unroll
    for (int i = 0; i < 2; ++i)
#pragma unroll
        for (int j = 0; j < 2; ++j)
#pragma unroll
            for (int r = 0; r < 8; ++r) {
                int row = rowbase + wrow * 32 + i * 16 + (hs ? 8 + r : r);
                int col = colbase + wcol * 32 + j * 16 + r15;
                float v = acc[i][j][r];
                if (col < 2048) {
                    qk[((size_t)m * ROWS_ + row) * QKST_ + col] = (bf16_t)v;
                } else {
                    int d = col - 2048;
                    int b = row >> 7, s = row & 127;
                    bf16_t* dst = (d < 512) ? vTr : vTi;
                    int dd = d & 511;
                    dst[(((size_t)(m * B_ + b) * D_) + dd) * S_ + s] = (bf16_t)v;
                }
            }
}

// ---------------------------------------------------------------------------
// Attention + ModReLU + ComplexLayerNorm, one workgroup per (m,b).
// ---------------------------------------------------------------------------
__global__ void __launch_bounds__(256) k_attn(const bf16_t* __restrict__ qk,
                                              const bf16_t* __restrict__ vTr,
                                              const bf16_t* __restrict__ vTi,
                                              const float* __restrict__ mod_bias,
                                              const float* __restrict__ ln_scale,
                                              const float* __restrict__ ln_shift,
                                              float* __restrict__ mr, float* __restrict__ mi) {
    __shared__ bf16_t Pl[128 * 128];                     // 32KB softmax probs
    int tid = threadIdx.x, wave = tid >> 5, lane = tid & 31;
    int r15 = lane & 15, hs = lane >> 4;
    int mb = blockIdx.x; int m = mb >> 3, b = mb & 7;
    const bf16_t* base = qk + ((size_t)m * ROWS_ + b * S_) * QKST_;
    int strip = wave * 16;                               // 16 query rows per wave

    // ---- scores = (qf @ kf^T) * D^-0.5 ----
    v8f sc[8];
#pragma unroll
    for (int nt = 0; nt < 8; ++nt) sc[nt] = (v8f){};
    for (int kt = 0; kt < 32; ++kt) {
        const bf16_t* qp = base + (size_t)(strip + r15) * QKST_ + kt * 32;
        v16bf a = mk16(*(const v8bf*)(qp + (hs ? 16 : 0)), *(const v8bf*)(qp + (hs ? 24 : 8)));
#pragma unroll
        for (int nt = 0; nt < 8; ++nt) {
            const bf16_t* kp = base + (size_t)(nt * 16 + r15) * QKST_ + 1024 + kt * 32 + (hs ? 16 : 0);
            v16bf bb = mk16(*(const v8bf*)kp, *(const v8bf*)(kp + 8));
            sc[nt] = wmma_bf16(a, bb, sc[nt]);
        }
    }
    const float scale = 0.044194173824159216f;           // 512^-0.5

    // ---- row-wise softmax (rows live in 16-lane halves of C layout) ----
#pragma unroll
    for (int r = 0; r < 8; ++r) {
        float mx = -3.0e38f;
#pragma unroll
        for (int nt = 0; nt < 8; ++nt) { float v = sc[nt][r] * scale; sc[nt][r] = v; mx = fmaxf(mx, v); }
#pragma unroll
        for (int msk = 1; msk < 16; msk <<= 1) mx = fmaxf(mx, __shfl_xor(mx, msk, 32));
        float sm = 0.f;
#pragma unroll
        for (int nt = 0; nt < 8; ++nt) { float e = __expf(sc[nt][r] - mx); sc[nt][r] = e; sm += e; }
#pragma unroll
        for (int msk = 1; msk < 16; msk <<= 1) sm += __shfl_xor(sm, msk, 32);
        float inv = 1.f / sm;
        int prow = strip + (hs ? 8 + r : r);
#pragma unroll
        for (int nt = 0; nt < 8; ++nt) Pl[prow * 128 + nt * 16 + r15] = (bf16_t)(sc[nt][r] * inv);
    }
    __syncthreads();

    // ---- out = P @ V (r,i) -> ModReLU -> row stats ----
    const bf16_t* vrb = vTr + (size_t)mb * D_ * S_;
    const bf16_t* vib = vTi + (size_t)mb * D_ * S_;
    float* mrb = mr + (size_t)mb * SD_;
    float* mib = mi + (size_t)mb * SD_;
    float ssum[8] = {}, ssq[8] = {};
    for (int nt = 0; nt < 32; ++nt) {
        int dcol = nt * 16 + r15;
        float bias = mod_bias[m * D_ + dcol];
        v8f ar = (v8f){}, ai = (v8f){};
#pragma unroll
        for (int ks = 0; ks < 4; ++ks) {
            const bf16_t* pp = &Pl[(strip + r15) * 128 + ks * 32];
            v16bf a = mk16(*(const v8bf*)(pp + (hs ? 16 : 0)), *(const v8bf*)(pp + (hs ? 24 : 8)));
            const bf16_t* vp = vrb + (size_t)dcol * S_ + ks * 32 + (hs ? 16 : 0);
            ar = wmma_bf16(a, mk16(*(const v8bf*)vp, *(const v8bf*)(vp + 8)), ar);
            const bf16_t* vp2 = vib + (size_t)dcol * S_ + ks * 32 + (hs ? 16 : 0);
            ai = wmma_bf16(a, mk16(*(const v8bf*)vp2, *(const v8bf*)(vp2 + 8)), ai);
        }
#pragma unroll
        for (int r = 0; r < 8; ++r) {
            float orv = ar[r], oiv = ai[r];
            float mag = sqrtf(orv * orv + oiv * oiv) + EPS_;
            float g = fmaxf(mag + bias, 0.f) / mag;      // ModReLU
            orv *= g; oiv *= g;
            float m2 = sqrtf(orv * orv + oiv * oiv) + EPS_;
            ssum[r] += m2; ssq[r] += m2 * m2;
            int s = strip + (hs ? 8 + r : r);
            size_t off = (size_t)s * D_ + dcol;
            mrb[off] = orv; mib[off] = oiv;              // raw post-ModReLU
        }
    }
    float mu[8], rstd[8];
#pragma unroll
    for (int r = 0; r < 8; ++r) {
        float s1 = ssum[r], s2 = ssq[r];
#pragma unroll
        for (int msk = 1; msk < 16; msk <<= 1) { s1 += __shfl_xor(s1, msk, 32); s2 += __shfl_xor(s2, msk, 32); }
        float mn = s1 * (1.f / (float)D_);
        float var = (s2 - (float)D_ * mn * mn) * (1.f / (float)(D_ - 1));  // unbiased
        mu[r] = mn; rstd[r] = rsqrtf(var + EPS_);
    }
    // ---- normalize magnitude, keep phase (same lane re-reads its own stores) ----
    for (int nt = 0; nt < 32; ++nt) {
        int dcol = nt * 16 + r15;
        float lsc = ln_scale[m * D_ + dcol], lsh = ln_shift[m * D_ + dcol];
#pragma unroll
        for (int r = 0; r < 8; ++r) {
            int s = strip + (hs ? 8 + r : r);
            size_t off = (size_t)s * D_ + dcol;
            float orv = mrb[off], oiv = mib[off];
            float m2 = sqrtf(orv * orv + oiv * oiv) + EPS_;
            float nm = (m2 - mu[r]) * rstd[r] * lsc + lsh;
            float r0 = sqrtf(orv * orv + oiv * oiv);
            float c, si;
            if (r0 > 0.f) { c = orv / r0; si = oiv / r0; } else { c = 1.f; si = 0.f; }
            mrb[off] = nm * c; mib[off] = nm * si;
        }
    }
}

// salience: sal[m,b] = mean_s(mr)·w_sal[:D] + mean_s(mi)·w_sal[D:]
__global__ void k_sal(const float* __restrict__ mr, const float* __restrict__ mi,
                      const float* __restrict__ w_sal, float* __restrict__ sal) {
    __shared__ float red[256];
    int mb = blockIdx.x; int m = mb >> 3;
    const float* mrb = mr + (size_t)mb * SD_;
    const float* mib = mi + (size_t)mb * SD_;
    const float* ws = w_sal + m * K2D_;
    float acc = 0.f;
    for (int i = threadIdx.x; i < SD_; i += 256) {
        int d = i & 511;
        acc += mrb[i] * ws[d] + mib[i] * ws[512 + d];
    }
    red[threadIdx.x] = acc; __syncthreads();
    for (int st = 128; st > 0; st >>= 1) {
        if (threadIdx.x < st) red[threadIdx.x] += red[threadIdx.x + st];
        __syncthreads();
    }
    if (threadIdx.x == 0) sal[mb] = red[0] * (1.f / (float)S_);
}

// module bias from controller state + module-competition softmax (TEMP=1)
__global__ void k_wcomp(const float* __restrict__ h, const float* __restrict__ bias_W,
                        const float* __restrict__ bias_b, const float* __restrict__ sal,
                        float* __restrict__ w) {
    __shared__ float mbv[B_][M_];
    int t = threadIdx.x;
    if (t < 32) {
        int b = t >> 2, mm = t & 3;
        float acc = bias_b[mm];
        for (int d = 0; d < D_; ++d) acc += h[b * D_ + d] * bias_W[d * M_ + mm];
        mbv[b][mm] = acc;
    }
    __syncthreads();
    if (t < B_) {
        int b = t;
        float l[M_], mx = -3e38f;
        for (int mm = 0; mm < M_; ++mm) { l[mm] = sal[mm * B_ + b] + mbv[b][mm]; mx = fmaxf(mx, l[mm]); }
        float s = 0.f;
        for (int mm = 0; mm < M_; ++mm) { l[mm] = __expf(l[mm] - mx); s += l[mm]; }
        for (int mm = 0; mm < M_; ++mm) w[mm * B_ + b] = l[mm] / s;
    }
}

// z = sum_m w[m,b]*m{r,i}; also refresh bf16 X for next step's QKV
__global__ void k_combine(const float* __restrict__ mr, const float* __restrict__ mi,
                          const float* __restrict__ w, float* __restrict__ zr,
                          float* __restrict__ zi, bf16_t* __restrict__ Xbf) {
    int idx = blockIdx.x * 256 + threadIdx.x;            // < 524288
    if (idx >= BSD_) return;
    int b = idx >> 16;
    int rem = idx & (SD_ - 1);
    float gr = 0.f, gi = 0.f;
#pragma unroll
    for (int m = 0; m < M_; ++m) {
        float wv = w[m * B_ + b];
        size_t off = (size_t)(m * B_ + b) * SD_ + rem;
        gr += wv * mr[off]; gi += wv * mi[off];
    }
    zr[idx] = gr; zi[idx] = gi;
    int row = idx >> 9, d = idx & 511;
    Xbf[(size_t)row * K2D_ + d]       = (bf16_t)gr;
    Xbf[(size_t)row * K2D_ + 512 + d] = (bf16_t)gi;
}

// ACT halting: p = sigmoid(gwp·w_halt + b); update cum; emit wt[b]
__global__ void k_halt(const float* __restrict__ pooled, const float* __restrict__ w_halt,
                       const float* __restrict__ b_halt, float* __restrict__ cum,
                       float* __restrict__ wt) {
    int t = threadIdx.x; int b = t >> 5, lane = t & 31;  // 8 waves, one per batch
    float acc = 0.f;
    for (int c = lane; c < K2D_; c += 32) acc += pooled[b * K2D_ + c] * w_halt[c];
#pragma unroll
    for (int msk = 16; msk > 0; msk >>= 1) acc += __shfl_xor(acc, msk, 32);
    if (lane == 0) {
        float p = sigmoidf_(acc + b_halt[0]);
        float cv = cum[b];
        float still = (cv < THR_) ? 1.f : 0.f;
        bool nh = ((cv + p) >= THR_) && (cv < THR_);
        float wv = (nh ? (1.f - cv) : p) * still;
        cum[b] = cv + wv;
        wt[b] = wv;
    }
}

// accumulate workspace: ar += wt*gr, ai += wt*gi  (d_out layout [2,B,S,D])
__global__ void k_act(const float* __restrict__ zr, const float* __restrict__ zi,
                      const float* __restrict__ wt, float* __restrict__ out) {
    int idx = blockIdx.x * 256 + threadIdx.x;            // < 524288
    if (idx >= BSD_) return;
    float wv = wt[idx >> 16];
    out[idx]        += wv * zr[idx];
    out[BSD_ + idx] += wv * zi[idx];
}

// ---------------------------------------------------------------------------
extern "C" void kernel_launch(void* const* d_in, const int* in_sizes, int n_in,
                              void* d_out, int out_size, void* d_ws, size_t ws_size,
                              hipStream_t stream) {
    (void)in_sizes; (void)n_in; (void)out_size; (void)ws_size;
    const float* x_real  = (const float*)d_in[0];
    const float* x_imag  = (const float*)d_in[1];
    const float* Wq_r    = (const float*)d_in[2];
    const float* Wq_i    = (const float*)d_in[3];
    const float* Wk_r    = (const float*)d_in[4];
    const float* Wk_i    = (const float*)d_in[5];
    const float* Wv_r    = (const float*)d_in[6];
    const float* Wv_i    = (const float*)d_in[7];
    const float* mod_bias= (const float*)d_in[8];
    const float* ln_scale= (const float*)d_in[9];
    const float* ln_shift= (const float*)d_in[10];
    const float* w_sal   = (const float*)d_in[11];
    const float* gru_Wih = (const float*)d_in[12];
    const float* gru_Whh = (const float*)d_in[13];
    const float* gru_bih = (const float*)d_in[14];
    const float* gru_bhh = (const float*)d_in[15];
    const float* bias_W  = (const float*)d_in[16];
    const float* bias_b  = (const float*)d_in[17];
    const float* w_halt  = (const float*)d_in[18];
    const float* b_halt  = (const float*)d_in[19];
    float* out = (float*)d_out;

    char* p = (char*)d_ws;
    auto carve = [&](size_t bytes) -> char* {
        char* r = p; p += (bytes + 255) & ~(size_t)255; return r;
    };
    float*  zr     = (float*)carve((size_t)BSD_ * 4);
    float*  zi     = (float*)carve((size_t)BSD_ * 4);
    bf16_t* Xbf    = (bf16_t*)carve((size_t)ROWS_ * K2D_ * 2);
    bf16_t* Wpack  = (bf16_t*)carve((size_t)M_ * 32 * 192 * 512 * 2);
    bf16_t* qk     = (bf16_t*)carve((size_t)M_ * ROWS_ * QKST_ * 2);
    bf16_t* vTr    = (bf16_t*)carve((size_t)M_ * B_ * D_ * S_ * 2);
    bf16_t* vTi    = (bf16_t*)carve((size_t)M_ * B_ * D_ * S_ * 2);
    float*  mr     = (float*)carve((size_t)M_ * B_ * SD_ * 4);
    float*  mi     = (float*)carve((size_t)M_ * B_ * SD_ * 4);
    float*  gx     = (float*)carve((size_t)B_ * 1536 * 4);
    float*  gh     = (float*)carve((size_t)B_ * 1536 * 4);
    float*  pooled = (float*)carve((size_t)B_ * K2D_ * 4);
    float*  h      = (float*)carve((size_t)B_ * D_ * 4);
    float*  sal    = (float*)carve((size_t)M_ * B_ * 4);
    float*  w      = (float*)carve((size_t)M_ * B_ * 4);
    float*  cum    = (float*)carve((size_t)B_ * 4);
    float*  wt     = (float*)carve((size_t)B_ * 4);

    // one-time per launch: pack weights, init state, first pooled
    k_pack<<<49152, 256, 0, stream>>>(Wq_r, Wq_i, Wk_r, Wk_i, Wv_r, Wv_i, Wpack);
    k_init<<<2048, 256, 0, stream>>>(x_real, x_imag, zr, zi, Xbf, out, h, cum);
    k_pool<<<32, 256, 0, stream>>>(zr, zi, pooled);

    for (int step = 0; step < DEPTH_; ++step) {
        k_gru_mm<<<96, 256, 0, stream>>>(pooled, h, gru_Wih, gru_Whh, gru_bih, gru_bhh, gx, gh);
        k_gru_gate<<<16, 256, 0, stream>>>(gx, gh, h);
        k_qkv<<<dim3(48, 8, 4), 256, 0, stream>>>(Xbf, Wpack, qk, vTr, vTi);
        k_attn<<<32, 256, 0, stream>>>(qk, vTr, vTi, mod_bias, ln_scale, ln_shift, mr, mi);
        k_sal<<<32, 256, 0, stream>>>(mr, mi, w_sal, sal);
        k_wcomp<<<1, 64, 0, stream>>>(h, bias_W, bias_b, sal, w);
        k_combine<<<2048, 256, 0, stream>>>(mr, mi, w, zr, zi, Xbf);
        k_pool<<<32, 256, 0, stream>>>(zr, zi, pooled);          // pooled == gwp
        k_halt<<<1, 256, 0, stream>>>(pooled, w_halt, b_halt, cum, wt);
        k_act<<<2048, 256, 0, stream>>>(zr, zi, wt, out);
    }
}
